// SwinDecoder_80900003988070
// MI455X (gfx1250) — compile-verified, matrix operand order checked by reading, not run on hardware
//
#include <hip/hip_runtime.h>

typedef unsigned short u16;
typedef __bf16 bf16_t;
typedef bf16_t v16bf __attribute__((ext_vector_type(16)));
typedef float  v8f   __attribute__((ext_vector_type(8)));
typedef unsigned int u32v4 __attribute__((ext_vector_type(4)));
typedef int i32v4 __attribute__((ext_vector_type(4)));
typedef int i32v8 __attribute__((ext_vector_type(8)));

#define TOKN   36864      // B*H*W tokens
#define CDIM   768
#define HIDDIM 3072
#define QKVDIM 2304

// ---------- helpers ----------
__device__ __forceinline__ u16 f2bf(float f){
  unsigned u = __float_as_uint(f);
  unsigned r = u + 0x7FFFu + ((u >> 16) & 1u);   // round-to-nearest-even
  return (u16)(r >> 16);
}
__device__ __forceinline__ float bf2f(u16 h){
  return __uint_as_float(((unsigned)h) << 16);
}

__device__ __forceinline__ v8f wmma_bf16(v16bf a, v16bf b, v8f c){
  return __builtin_amdgcn_wmma_f32_16x16x32_bf16(false, a, false, b, (short)0, c, false, false);
}

// Per-lane bf16 A/B fragment from an LDS tile (row-major, ldk halves/row).
__device__ __forceinline__ v16bf load_frag(const u16* tile, int row, int ldk, int k0){
  int lane = threadIdx.x & 31;
  int rr = row + (lane & 15);
  int kb = k0 + ((lane & 16) ? 8 : 0);
  const u16* p = tile + rr * ldk + kb;
  union { uint4 u[2]; v16bf v; } cv;
  cv.u[0] = *(const uint4*)p;
  cv.u[1] = *(const uint4*)(p + 16);
  return cv.v;
}

// ---------- Tensor Data Mover: 2D tile (rows x row_elems bf16) -> LDS ----------
// D# per CDNA5 ISA: group0 = {count/lds_addr/global_addr/type},
// group1 = {data_size, pad_enable, pad_interval, pad_amount, tensor dims, tile dims, strides}.
// pad_interval_code: DWORDs before pad (3 => 16 DW = 64B, 4 => 32 DW = 128B)
// pad_amount_code:   pad DWORDs (3 => 4 DW = 16B)
__device__ __forceinline__ void tdm_load_2d(const void* gptr, unsigned lds_off,
                                            int rows, int row_elems, int pitch_elems,
                                            int pad_interval_code, int pad_amount_code)
{
  unsigned long long ga = (unsigned long long)gptr;
  u32v4 g0;
  g0[0] = 1u;                                                   // count=1 (valid user D#)
  g0[1] = lds_off;                                              // LDS byte address
  g0[2] = (unsigned)ga;                                         // global addr [31:0]
  g0[3] = (unsigned)((ga >> 32) & 0x01FFFFFFu) | (2u << 30);    // addr[56:32] | type=2
  i32v8 g1;
  g1[0] = (int)((1u << 16) |                                    // data_size = 2 bytes
                (1u << 20) |                                    // pad_enable
                ((unsigned)pad_interval_code << 22) |
                ((unsigned)pad_amount_code << 25));
  unsigned td0 = (unsigned)pitch_elems;                         // tensor_dim0 (elems)
  unsigned td1 = 0x00100000u;                                   // tensor_dim1 (large)
  g1[1] = (int)((td0 & 0xFFFFu) << 16);                         // tensor_dim0 lo16 @bit48
  g1[2] = (int)((td0 >> 16) | ((td1 & 0xFFFFu) << 16));         // tensor_dim0 hi / dim1 lo
  g1[3] = (int)((td1 >> 16) | ((unsigned)row_elems << 16));     // dim1 hi | tile_dim0
  g1[4] = (int)((unsigned)rows & 0xFFFFu);                      // tile_dim1 (tile_dim2=0)
  g1[5] = (int)(unsigned)pitch_elems;                           // tensor_dim0_stride lo32
  g1[6] = 0;                                                    // stride hi16 | dim1 stride lo
  g1[7] = 0;
  i32v4 z4 = {0, 0, 0, 0};
#if __clang_major__ >= 23
  i32v8 z8 = {0, 0, 0, 0, 0, 0, 0, 0};
  __builtin_amdgcn_tensor_load_to_lds(g0, g1, z4, z4, z8, 0);
#else
  __builtin_amdgcn_tensor_load_to_lds(g0, g1, z4, z4, 0);
#endif
}

// window-order row m -> image-order row index, applying the (+4,+4) mod shift.
__device__ __forceinline__ size_t swin_map_row(size_t m){
  int n     = (int)(m & 63);
  size_t b_ = m >> 6;
  int b  = (int)(b_ / 288);
  int wi = (int)(b_ % 288);
  int wh = wi / 24, ww = wi % 24;
  int r = n >> 3, s = n & 7;
  int hh = (wh * 8 + r + 4) % 96;
  int w2 = (ww * 8 + s + 4) % 192;
  return (size_t)b * (96 * 192) + (size_t)hh * 192 + (size_t)w2;
}

// ---------- elementwise kernels ----------
__global__ __launch_bounds__(256) void f2bf_kernel(const float* __restrict__ s,
                                                   u16* __restrict__ d, size_t n){
  size_t i = ((size_t)blockIdx.x * 256 + threadIdx.x) * 4;
  if (i + 3 < n){
    float4 v = *(const float4*)(s + i);
    u16 o[4] = { f2bf(v.x), f2bf(v.y), f2bf(v.z), f2bf(v.w) };
    *(uint2*)(d + i) = *(const uint2*)o;
  }
}

// roll(-4,-4) + window partition + f32->bf16.  One thread per 8-channel chunk.
__global__ __launch_bounds__(256) void partition_shift_kernel(const float* __restrict__ x,
                                                              u16* __restrict__ xw){
  size_t tid = (size_t)blockIdx.x * 256 + threadIdx.x;
  size_t m = tid / 96;
  int   c8 = (int)(tid % 96);
  size_t tm = swin_map_row(m);
  const float* src = x + tm * CDIM + (size_t)c8 * 8;
  float4 a = *(const float4*)src;
  float4 b = *(const float4*)(src + 4);
  u16 o[8] = { f2bf(a.x), f2bf(a.y), f2bf(a.z), f2bf(a.w),
               f2bf(b.x), f2bf(b.y), f2bf(b.z), f2bf(b.w) };
  *(uint4*)(xw + m * CDIM + (size_t)c8 * 8) = *(const uint4*)o;
}

// 2D RoPE on q,k (+ fold softmax scale into q); one thread per rotation pair.
__global__ __launch_bounds__(256) void rope_kernel(const u16* __restrict__ qkv,
                                                   u16* __restrict__ qo,
                                                   u16* __restrict__ ko){
  size_t tid = (size_t)blockIdx.x * 256 + threadIdx.x;
  int t = (int)(tid & 31);
  size_t th = tid >> 5;
  int h = (int)(th % 12);
  size_t tok = th / 12;
  int n = (int)(tok & 63);
  int r = n >> 3, s = n & 7;
  int pos, f, d0;
  if (t < 16){ pos = r; f = t;      d0 = 2 * t; }
  else       { pos = s; f = t - 16; d0 = 32 + 2 * (t - 16); }
  float inv = __expf(-(float)f * (9.210340371976184f / 16.0f));   // 10000^(-f/16)
  float ang = (float)pos * inv;
  float c = __cosf(ang), sn = __sinf(ang);
  size_t qbase = tok * QKVDIM + (size_t)h * 64;
  size_t kbase = qbase + CDIM;
  float q1 = bf2f(qkv[qbase + d0]), q2 = bf2f(qkv[qbase + d0 + 1]);
  float k1 = bf2f(qkv[kbase + d0]), k2 = bf2f(qkv[kbase + d0 + 1]);
  size_t b_ = tok >> 6;
  size_t ob = (((b_ * 12 + h) * 64) + n) * 64;
  const float SC = 0.125f;               // 64^-0.5
  qo[ob + d0]     = f2bf((q1 * c - q2 * sn) * SC);
  qo[ob + d0 + 1] = f2bf((q1 * sn + q2 * c) * SC);
  ko[ob + d0]     = f2bf(k1 * c - k2 * sn);
  ko[ob + d0 + 1] = f2bf(k1 * sn + k2 * c);
}

// v slice of qkv -> [b_, h, n, d]; one thread per 8-wide chunk.
__global__ __launch_bounds__(256) void vperm_kernel(const u16* __restrict__ qkv,
                                                    u16* __restrict__ vo){
  size_t tid = (size_t)blockIdx.x * 256 + threadIdx.x;
  int c8 = (int)(tid & 7);
  size_t th = tid >> 3;
  int h = (int)(th % 12);
  size_t tok = th / 12;
  size_t b_ = tok >> 6;
  int n = (int)(tok & 63);
  size_t src = tok * QKVDIM + 2 * CDIM + (size_t)h * 64 + (size_t)c8 * 8;
  size_t dst = (((b_ * 12 + h) * 64) + n) * 64 + (size_t)c8 * 8;
  *(uint4*)(vo + dst) = *(const uint4*)(qkv + src);
}

// ---------- tiled WMMA GEMM with TDM double-buffered tile staging ----------
// out = A(bf16 MxK) @ Bw(bf16 NxK)^T + bias
// EPI 0: bf16 store  EPI 1: relu->bf16  EPI 2: scatter(map)+resid->f32  EPI 3: +resid->f32
template<int EPI>
__global__ __launch_bounds__(256)
void gemm_bf16_kernel(const u16* __restrict__ A, const u16* __restrict__ Bw,
                      const float* __restrict__ bias,
                      u16* __restrict__ outb, float* __restrict__ outf,
                      const float* __restrict__ resid,
                      int M, int N, int K)
{
  constexpr int LDT = 40;                       // 64B row + 16B TDM pad = 80B pitch
  constexpr int AH = 128 * LDT;                 // A tile halves
  constexpr int BH = 256 * LDT;                 // B tile halves
  constexpr int BUFH = AH + BH;                 // halves per double-buffer slot
  extern __shared__ u16 smem[];                 // 2*BUFH halves = 61440 B
  const unsigned lds_base = (unsigned)(unsigned long long)(uintptr_t)smem; // addr[31:0]=LDS offset

  const int mblk = blockIdx.y * 128;
  const int nblk = blockIdx.x * 256;
  const int wv   = __builtin_amdgcn_readfirstlane((int)(threadIdx.x >> 5)); // scalar wave id
  const int wm0  = (wv & 1) * 64;
  const int wn0  = (wv >> 1) * 64;

  // each wave DMAs its own A slice (16 rows) and B slice (32 rows) per K-step
  auto issue = [&](int k0, int buf){
    const u16* ga = &A[(size_t)(mblk + wv * 16) * K + k0];
    tdm_load_2d(ga, lds_base + (unsigned)(buf * BUFH + wv * 16 * LDT) * 2u,
                16, 32, K, /*pad: 16DW*/3, /*4DW*/3);
    const u16* gb = &Bw[(size_t)(nblk + wv * 32) * K + k0];
    tdm_load_2d(gb, lds_base + (unsigned)(buf * BUFH + AH + wv * 32 * LDT) * 2u,
                32, 32, K, 3, 3);
  };

  v8f acc[4][4] = {};
  issue(0, 0);
  int buf = 0;
  for (int k0 = 0; k0 < K; k0 += 32){
    if (k0 + 32 < K){
      issue(k0 + 32, buf ^ 1);                       // prefetch next buffer
      __builtin_amdgcn_s_wait_tensorcnt((short)2);   // oldest 2 (current buf) complete
    } else {
      __builtin_amdgcn_s_wait_tensorcnt((short)0);
    }
    __syncthreads();
    const u16* As = smem + buf * BUFH;
    const u16* Bs = As + AH;
    v16bf fa[4], fb[4];
    #pragma unroll
    for (int i = 0; i < 4; i++) fa[i] = load_frag(As, wm0 + i * 16, LDT, 0);
    #pragma unroll
    for (int j = 0; j < 4; j++) fb[j] = load_frag(Bs, wn0 + j * 16, LDT, 0);
    #pragma unroll
    for (int i = 0; i < 4; i++)
      #pragma unroll
      for (int j = 0; j < 4; j++)
        acc[i][j] = wmma_bf16(fa[i], fb[j], acc[i][j]);
    __syncthreads();
    buf ^= 1;
  }

  const int lane = threadIdx.x & 31;
  const int lcol = lane & 15;
  const int lrow = (lane & 16) ? 8 : 0;
  #pragma unroll
  for (int i = 0; i < 4; i++){
    #pragma unroll
    for (int j = 0; j < 4; j++){
      const int n = nblk + wn0 + j * 16 + lcol;
      const float bv = bias[n];
      #pragma unroll
      for (int r = 0; r < 8; r++){
        const int m = mblk + wm0 + i * 16 + lrow + r;
        float v = acc[i][j][r] + bv;
        if (EPI == 0){
          outb[(size_t)m * N + n] = f2bf(v);
        } else if (EPI == 1){
          outb[(size_t)m * N + n] = f2bf(v > 0.f ? v : 0.f);
        } else if (EPI == 2){
          size_t tm = swin_map_row((size_t)m);
          outf[tm * N + n] = resid[tm * N + n] + v;
        } else {
          size_t mm = (size_t)m * N + n;
          outf[mm] = resid[mm] + v;
        }
      }
    }
  }
}

// ---------- windowed attention: one wave per (window, head) ----------
// LDS layout (dynamic): Qs[64*72]h | Ks[64*72]h | VT[64*72]h | Ps[64*72]h | Ss[64*65]f
#define ATTN_LDS_BYTES (4 * 64 * 72 * 2 + 64 * 65 * 4)
__global__ __launch_bounds__(32)
void attn_kernel(const u16* __restrict__ q, const u16* __restrict__ k,
                 const u16* __restrict__ v, const float* __restrict__ mask,
                 u16* __restrict__ ao)
{
  extern __shared__ char asmem[];
  u16*  Qs = (u16*)(asmem);
  u16*  Ks = (u16*)(asmem + 9216);
  u16*  VT = (u16*)(asmem + 18432);
  u16*  Ps = (u16*)(asmem + 27648);
  float* Ss = (float*)(asmem + 36864);
  const unsigned lds_base = (unsigned)(unsigned long long)(uintptr_t)asmem;

  const int bh = blockIdx.x;
  const int b_ = bh / 12;
  const int h  = bh % 12;
  const int wi = b_ % 288;
  const int lane = threadIdx.x;
  const size_t base = (size_t)bh * 64 * 64;

  // Q,K tiles via Tensor Data Mover (pad 128B rows -> 144B pitch = 72 halves)
  tdm_load_2d(q + base, lds_base,         64, 64, 64, /*32DW*/4, /*4DW*/3);
  tdm_load_2d(k + base, lds_base + 9216u, 64, 64, 64, 4, 3);

  // V loaded manually (transposed into VT)
  for (int c = lane; c < 512; c += 32){
    int r = c >> 3, kk = (c & 7) * 8;
    uint4 vv = *(const uint4*)&v[base + (size_t)r * 64 + kk];
    const u16* ve = (const u16*)&vv;
    #pragma unroll
    for (int e = 0; e < 8; e++) VT[(kk + e) * 72 + r] = ve[e];
  }
  __builtin_amdgcn_s_wait_tensorcnt((short)0);
  __syncthreads();

  // S = Q K^T   (scale already folded into Q)
  #pragma unroll
  for (int ti = 0; ti < 4; ti++){
    #pragma unroll
    for (int tj = 0; tj < 4; tj++){
      v8f acc = {};
      #pragma unroll
      for (int ks = 0; ks < 2; ks++){
        v16bf fa = load_frag(Qs, ti * 16, 72, ks * 32);
        v16bf fb = load_frag(Ks, tj * 16, 72, ks * 32);
        acc = wmma_bf16(fa, fb, acc);
      }
      int col = tj * 16 + (lane & 15);
      int r0  = ti * 16 + ((lane & 16) ? 8 : 0);
      #pragma unroll
      for (int r = 0; r < 8; r++) Ss[(r0 + r) * 65 + col] = acc[r];
    }
  }
  __syncthreads();

  // masked softmax over rows; each lane handles 2 rows
  #pragma unroll
  for (int rr = 0; rr < 2; rr++){
    int n = lane + rr * 32;
    const float* mrow = mask + ((size_t)wi * 64 + n) * 64;
    float tv[64];
    float mx = -1e30f;
    #pragma unroll
    for (int m2 = 0; m2 < 64; m2++){
      float t = Ss[n * 65 + m2] + mrow[m2];
      tv[m2] = t;
      mx = t > mx ? t : mx;
    }
    float sum = 0.f;
    #pragma unroll
    for (int m2 = 0; m2 < 64; m2++){ float e = __expf(tv[m2] - mx); tv[m2] = e; sum += e; }
    float inv = 1.f / sum;
    #pragma unroll
    for (int m2 = 0; m2 < 64; m2++) Ps[n * 72 + m2] = f2bf(tv[m2] * inv);
  }
  __syncthreads();

  // O = P V   (B = VT rows = V columns)
  #pragma unroll
  for (int ti = 0; ti < 4; ti++){
    #pragma unroll
    for (int tj = 0; tj < 4; tj++){
      v8f acc = {};
      #pragma unroll
      for (int ks = 0; ks < 2; ks++){
        v16bf fa = load_frag(Ps, ti * 16, 72, ks * 32);
        v16bf fb = load_frag(VT, tj * 16, 72, ks * 32);
        acc = wmma_bf16(fa, fb, acc);
      }
      int dcol = tj * 16 + (lane & 15);
      int r0   = ti * 16 + ((lane & 16) ? 8 : 0);
      #pragma unroll
      for (int r = 0; r < 8; r++){
        size_t row = (size_t)b_ * 64 + r0 + r;
        ao[row * CDIM + (size_t)h * 64 + dcol] = f2bf(acc[r]);
      }
    }
  }
}

// ---------- LayerNorm over C=768, one block per row ----------
__global__ __launch_bounds__(256)
void ln_kernel(const float* __restrict__ in, const float* __restrict__ g,
               const float* __restrict__ b, float* __restrict__ outf,
               u16* __restrict__ outb)
{
  __shared__ float red[256];
  const size_t row = blockIdx.x;
  const float* x = in + row * CDIM;
  const int t = threadIdx.x;
  float v0 = x[t], v1 = x[t + 256], v2 = x[t + 512];
  red[t] = v0 + v1 + v2;
  __syncthreads();
  for (int o = 128; o > 0; o >>= 1){ if (t < o) red[t] += red[t + o]; __syncthreads(); }
  float mean = red[0] * (1.f / 768.f);
  __syncthreads();
  float d0 = v0 - mean, d1 = v1 - mean, d2 = v2 - mean;
  red[t] = d0 * d0 + d1 * d1 + d2 * d2;
  __syncthreads();
  for (int o = 128; o > 0; o >>= 1){ if (t < o) red[t] += red[t + o]; __syncthreads(); }
  float rstd = rsqrtf(red[0] * (1.f / 768.f) + 1e-5f);
  float vv[3] = { d0, d1, d2 };
  #pragma unroll
  for (int e = 0; e < 3; e++){
    int c = t + e * 256;
    float val = vv[e] * rstd * g[c] + b[c];
    if (outf) outf[row * CDIM + c] = val;
    if (outb) outb[row * CDIM + c] = f2bf(val);
  }
}

// ---------- launcher ----------
extern "C" void kernel_launch(void* const* d_in, const int* in_sizes, int n_in,
                              void* d_out, int out_size, void* d_ws, size_t ws_size,
                              hipStream_t stream) {
  const float* x      = (const float*)d_in[0];
  const float* mask   = (const float*)d_in[1];
  const float* w_qkv  = (const float*)d_in[2];
  const float* b_qkv  = (const float*)d_in[3];
  const float* w_proj = (const float*)d_in[4];
  const float* b_proj = (const float*)d_in[5];
  const float* g1     = (const float*)d_in[6];
  const float* b1     = (const float*)d_in[7];
  const float* g2     = (const float*)d_in[8];
  const float* b2     = (const float*)d_in[9];
  const float* w_fc1  = (const float*)d_in[10];
  const float* b_fc1  = (const float*)d_in[11];
  const float* w_fc2  = (const float*)d_in[12];
  const float* b_fc2  = (const float*)d_in[13];
  float* out = (float*)d_out;

  const size_t TOKC2 = (size_t)TOKN * CDIM * 2;
  const size_t TOKC4 = (size_t)TOKN * CDIM * 4;
  const int GEMM_LDS = 2 * (128 * 40 + 256 * 40) * 2;   // 61440 B

  char* p = (char*)d_ws;
  u16* wqkv_bf  = (u16*)p; p += (size_t)QKVDIM * CDIM * 2;
  u16* wproj_bf = (u16*)p; p += (size_t)CDIM * CDIM * 2;
  u16* wfc1_bf  = (u16*)p; p += (size_t)HIDDIM * CDIM * 2;
  u16* wfc2_bf  = (u16*)p; p += (size_t)CDIM * HIDDIM * 2;
  char* xw_p  = p;  u16* xw_bf  = (u16*)p; p += TOKC2;
  char* qkv_p = p;  u16* qkv_bf = (u16*)p; p += (size_t)TOKN * QKVDIM * 2;
  u16* q_bf = (u16*)p; p += TOKC2;
  u16* k_bf = (u16*)p; p += TOKC2;
  u16* v_bf = (u16*)p; p += TOKC2;
  float* x1pre = (float*)p; p += TOKC4;   // later reused as y_pre
  float* x1f   = (float*)p; p += TOKC4;
  u16*   x1bf  = (u16*)p;  p += TOKC2;
  u16*   h_bf  = (u16*)qkv_p;             // overlays qkv+q (exactly TOKN*HIDDIM*2)
  u16*   ao_bf = (u16*)xw_p;              // overlays xw (consumed by QKV gemm)
  float* ypre  = x1pre;

  // 1) weights fp32 -> bf16
  f2bf_kernel<<<(QKVDIM * CDIM / 4 + 255) / 256, 256, 0, stream>>>(w_qkv,  wqkv_bf,  (size_t)QKVDIM * CDIM);
  f2bf_kernel<<<(CDIM * CDIM / 4 + 255) / 256, 256, 0, stream>>>(w_proj, wproj_bf, (size_t)CDIM * CDIM);
  f2bf_kernel<<<(HIDDIM * CDIM / 4 + 255) / 256, 256, 0, stream>>>(w_fc1, wfc1_bf, (size_t)HIDDIM * CDIM);
  f2bf_kernel<<<(CDIM * HIDDIM / 4 + 255) / 256, 256, 0, stream>>>(w_fc2, wfc2_bf, (size_t)CDIM * HIDDIM);

  // 2) shift + partition + convert
  partition_shift_kernel<<<(TOKN * 96) / 256, 256, 0, stream>>>(x, xw_bf);

  // 3) QKV GEMM
  gemm_bf16_kernel<0><<<dim3(QKVDIM / 256, TOKN / 128), 256, GEMM_LDS, stream>>>(
      xw_bf, wqkv_bf, b_qkv, qkv_bf, nullptr, nullptr, TOKN, QKVDIM, CDIM);

  // 4) RoPE(q,k)+scale, permute v
  rope_kernel<<<((size_t)TOKN * 12 * 32) / 256, 256, 0, stream>>>(qkv_bf, q_bf, k_bf);
  vperm_kernel<<<((size_t)TOKN * 12 * 8) / 256, 256, 0, stream>>>(qkv_bf, v_bf);

  // 5) windowed attention
  attn_kernel<<<576 * 12, 32, ATTN_LDS_BYTES, stream>>>(q_bf, k_bf, v_bf, mask, ao_bf);

  // 6) proj GEMM + reverse-shift scatter + residual -> x1_pre (f32)
  gemm_bf16_kernel<2><<<dim3(CDIM / 256, TOKN / 128), 256, GEMM_LDS, stream>>>(
      ao_bf, wproj_bf, b_proj, nullptr, x1pre, x, TOKN, CDIM, CDIM);

  // 7) LN1 -> x1 (f32 + bf16)
  ln_kernel<<<TOKN, 256, 0, stream>>>(x1pre, g1, b1, x1f, x1bf);

  // 8) FC1 + ReLU -> h (bf16)
  gemm_bf16_kernel<1><<<dim3(HIDDIM / 256, TOKN / 128), 256, GEMM_LDS, stream>>>(
      x1bf, wfc1_bf, b_fc1, h_bf, nullptr, nullptr, TOKN, HIDDIM, CDIM);

  // 9) FC2 + residual -> y_pre (f32)
  gemm_bf16_kernel<3><<<dim3(CDIM / 256, TOKN / 128), 256, GEMM_LDS, stream>>>(
      h_bf, wfc2_bf, b_fc2, nullptr, ypre, x1f, TOKN, CDIM, HIDDIM);

  // 10) LN2 -> output
  ln_kernel<<<TOKN, 256, 0, stream>>>(ypre, g2, b2, out, nullptr);
}
